// SparseSDTransformerLayer_86861418594904
// MI455X (gfx1250) — matrix operand order repaired and checked by reading, not compiled, not run on hardware
//
#include <hip/hip_runtime.h>
#include <hip/hip_bf16.h>

typedef __attribute__((ext_vector_type(16))) _Float16 v16h;
typedef __attribute__((ext_vector_type(8)))  _Float16 v8h;
typedef __attribute__((ext_vector_type(4)))  _Float16 v4h;
typedef __attribute__((ext_vector_type(8)))  float    v8f;
typedef __attribute__((ext_vector_type(4)))  float    v4f;

#define B_ 8
#define N_ 2048
#define C_ 64
#define H_ 2
#define D_ 32
#define E_ 65536
#define SCALE 0.125f          // C^-0.5 = 64^-0.5
#define SWPITCH 2056          // LDS row pitch in halves (multiple of 8 -> 16B aligned chunks)
#define NEGH (-60000.0f)      // f16-representable "-inf" for edge mask

static __device__ __forceinline__ v16h cat16(v8h lo, v8h hi) {
  return __builtin_shufflevector(lo, hi, 0,1,2,3,4,5,6,7,8,9,10,11,12,13,14,15);
}

// ---------------------------------------------------------------------------
// Projections: q/k/v (dyn) and qs/ks/vs (static), f32 accumulate -> f16 store.
// V stored transposed [b, c, n] so attn*V B-operands are contiguous.
// Also writes resid = x.
// ---------------------------------------------------------------------------
__global__ __launch_bounds__(256) void proj_kernel(
    const float* __restrict__ x,
    const float* __restrict__ wq, const float* __restrict__ wk,
    const float* __restrict__ wv, const float* __restrict__ bv,
    const float* __restrict__ wqs, const float* __restrict__ wks,
    const float* __restrict__ wvs, const float* __restrict__ bvs,
    float* __restrict__ resid,
    _Float16* __restrict__ q16, _Float16* __restrict__ k16, _Float16* __restrict__ vT16,
    _Float16* __restrict__ qs16, _Float16* __restrict__ ks16, _Float16* __restrict__ vsT16)
{
  extern __shared__ char smraw[];
  float* sm   = (float*)smraw;
  float* lwq  = sm;           float* lwk  = lwq  + 4096;
  float* lwv  = lwk  + 4096;  float* lwqs = lwv  + 4096;
  float* lwks = lwqs + 4096;  float* lwvs = lwks + 4096;
  float* lbv  = lwvs + 4096;  float* lbvs = lbv + 64;
  float* xb   = lbvs + 64;    // 256 floats = 4 rows

  int tid = threadIdx.x;
  for (int i = tid; i < 4096; i += 256) {
    lwq[i]  = wq[i];  lwk[i]  = wk[i];  lwv[i]  = wv[i];
    lwqs[i] = wqs[i]; lwks[i] = wks[i]; lwvs[i] = wvs[i];
  }
  if (tid < 64) { lbv[tid] = bv[tid]; lbvs[tid] = bvs[tid]; }
  __syncthreads();

  int r = tid >> 6, j = tid & 63;
  for (int it = 0; it < 16; ++it) {
    int rowbase = blockIdx.x * 64 + it * 4;
    xb[tid] = x[(long)rowbase * 64 + tid];
    __syncthreads();
    int row = rowbase + r;
    float aq = 0.f, ak = 0.f, av = lbv[j], aqs = 0.f, aks = 0.f, avs = lbvs[j];
    #pragma unroll 8
    for (int c = 0; c < 64; ++c) {
      float xa = xb[r * 64 + c];
      aq  = fmaf(xa, lwq [c * 64 + j], aq);
      ak  = fmaf(xa, lwk [c * 64 + j], ak);
      av  = fmaf(xa, lwv [c * 64 + j], av);
      aqs = fmaf(xa, lwqs[c * 64 + j], aqs);
      aks = fmaf(xa, lwks[c * 64 + j], aks);
      avs = fmaf(xa, lwvs[c * 64 + j], avs);
    }
    int b = row >> 11, n = row & 2047;
    long ridx = (long)row * 64 + j;
    q16 [ridx] = (_Float16)aq;
    k16 [ridx] = (_Float16)ak;
    qs16[ridx] = (_Float16)aqs;
    ks16[ridx] = (_Float16)aks;
    vT16 [(long)(b * 64 + j) * N_ + n] = (_Float16)av;
    vsT16[(long)(b * 64 + j) * N_ + n] = (_Float16)avs;
    resid[ridx] = xb[r * 64 + j];
    __syncthreads();
  }
}

// ---------------------------------------------------------------------------
// Edge bitmask build
// ---------------------------------------------------------------------------
__global__ __launch_bounds__(256) void mask_clear_kernel(unsigned* __restrict__ mask) {
  mask[blockIdx.x * 256 + threadIdx.x] = 0u;
}
__global__ __launch_bounds__(256) void mask_set_kernel(const int* __restrict__ edge,
                                                       unsigned* __restrict__ mask) {
  int e = blockIdx.x * 256 + threadIdx.x;
  int n = edge[e];
  int m = edge[E_ + e];
  atomicOr(&mask[n * 64 + (m >> 5)], 1u << (m & 31));
}

// ---------------------------------------------------------------------------
// Dynamic mean-threshold masked attention. One block = (b, h, 16-row tile).
// ---------------------------------------------------------------------------
__global__ __launch_bounds__(256) void dyn_attn_kernel(
    const _Float16* __restrict__ q16, const _Float16* __restrict__ k16,
    const _Float16* __restrict__ vT16,
    float* __restrict__ resid, float* __restrict__ attn)
{
  extern __shared__ char smraw[];
  _Float16* sw   = (_Float16*)smraw;                       // 16 x SWPITCH halves
  float* part    = (float*)(smraw + 16 * SWPITCH * 2);     // 8 * 256
  float* psum    = part + 8 * 256;                          // 256
  float* pmax    = psum + 256;                              // 256
  float* rowmax  = pmax + 256;                              // 16
  float* rowmean = rowmax + 16;                             // 16
  float* rowinv  = rowmean + 16;                            // 16

  int tid = threadIdx.x, lane = tid & 31, wv = tid >> 5;
  int blk = blockIdx.x;
  int tile = blk & 127, hb = blk >> 7, h = hb & 1, b = hb >> 1;
  int n0 = tile * 16;
  int m = lane & 15;

  // ---- Phase 1: scores (scaled) into LDS f16 via WMMA ----
  long qbase = ((long)(b * N_ + n0 + m)) * C_ + h * D_;
  int aoff = (lane < 16) ? 0 : 16;  // A: K 0..15 vs 16..31
  int boff = (lane < 16) ? 0 : 16;  // B: K 0..15 vs 16..31
  v16h afrag = cat16(*(const v8h*)(q16 + qbase + aoff),
                     *(const v8h*)(q16 + qbase + aoff + 8));
  for (int t = wv; t < 128; t += 8) {
    int c0 = t * 16;
    long kbase = ((long)(b * N_ + c0 + m)) * C_ + h * D_;
    v16h bfrag = cat16(*(const v8h*)(k16 + kbase + boff),
                       *(const v8h*)(k16 + kbase + boff + 8));
    v8f acc = {};
    acc = __builtin_amdgcn_wmma_f32_16x16x32_f16(false, afrag, false, bfrag,
                                                 (short)0, acc, false, false);
    #pragma unroll
    for (int r = 0; r < 8; ++r) {
      int row = (lane < 16) ? r : r + 8;
      sw[row * SWPITCH + c0 + m] = (_Float16)(acc[r] * SCALE);
    }
  }
  __syncthreads();

  // ---- Phase 2: per-row sum (mean) and max; vectorized, bank-swizzled ----
  int rowId = tid >> 4, sub = tid & 15;
  {
    const _Float16* rp = sw + rowId * SWPITCH + sub * 128;
    float mx = -3.0e38f, sm2 = 0.f;
    #pragma unroll 4
    for (int kk = 0; kk < 16; ++kk) {
      int k2 = (kk + sub) & 15;
      v8h hv = *(const v8h*)(rp + k2 * 8);
      #pragma unroll
      for (int e = 0; e < 8; ++e) {
        float sv = (float)hv[e];
        sm2 += sv; mx = fmaxf(mx, sv);
      }
    }
    psum[tid] = sm2; pmax[tid] = mx;
  }
  __syncthreads();
  if (tid < 16) {
    float mx = -3.0e38f, sm2 = 0.f;
    for (int k2 = 0; k2 < 16; ++k2) { sm2 += psum[tid * 16 + k2]; mx = fmaxf(mx, pmax[tid * 16 + k2]); }
    rowmax[tid] = mx; rowmean[tid] = sm2 * (1.0f / (float)N_);
  }
  __syncthreads();

  // ---- Phase 3: w = (s>=mean) ? exp(s-max) : 0 ; denom ----
  {
    _Float16* rp = sw + rowId * SWPITCH + sub * 128;
    float mean = rowmean[rowId], mxv = rowmax[rowId], dn = 0.f;
    #pragma unroll 4
    for (int kk = 0; kk < 16; ++kk) {
      int k2 = (kk + sub) & 15;
      v8h hv = *(const v8h*)(rp + k2 * 8);
      v8h ov;
      #pragma unroll
      for (int e = 0; e < 8; ++e) {
        float sv = (float)hv[e];
        float wt = (sv >= mean) ? __expf(sv - mxv) : 0.0f;
        ov[e] = (_Float16)wt;
        dn += wt;
      }
      *(v8h*)(rp + k2 * 8) = ov;
    }
    psum[tid] = dn;
  }
  __syncthreads();
  if (tid < 16) {
    float dn = 0.f;
    for (int k2 = 0; k2 < 16; ++k2) dn += psum[tid * 16 + k2];
    rowinv[tid] = 1.0f / dn;
  }
  __syncthreads();

  // ---- Phase 4: stream attn_d to global; b128 non-temporal stores ----
  long abase = (((long)(b * H_ + h)) * N_ + n0) * N_;
  for (int idx = tid * 4; idx < 16 * N_; idx += 1024) {
    int row = idx >> 11, col = idx & 2047;
    v4h hv = *(const v4h*)(sw + row * SWPITCH + col);
    float inv = rowinv[row];
    v4f o;
    o.x = (float)hv.x * inv; o.y = (float)hv.y * inv;
    o.z = (float)hv.z * inv; o.w = (float)hv.w * inv;
    __builtin_nontemporal_store(o, (v4f*)(attn + abase + (long)row * N_ + col));
  }

  // ---- Phase 5: out = (w * V) / denom, WMMA over K=2048, resid += ----
  {
    int nt = wv & 1, kq = wv >> 1;
    int c0 = nt * 16;
    long vbase = ((long)(b * C_) + h * D_ + c0 + m) * N_;
    v8f acc = {};
    for (int kc = kq * 512; kc < kq * 512 + 512; kc += 32) {
      v16h af = cat16(*(const v8h*)(sw + m * SWPITCH + kc + aoff),
                      *(const v8h*)(sw + m * SWPITCH + kc + aoff + 8));
      v16h bf = cat16(*(const v8h*)(vT16 + vbase + kc + boff),
                      *(const v8h*)(vT16 + vbase + kc + boff + 8));
      acc = __builtin_amdgcn_wmma_f32_16x16x32_f16(false, af, false, bf,
                                                   (short)0, acc, false, false);
    }
    #pragma unroll
    for (int r = 0; r < 8; ++r) part[wv * 256 + r * 32 + lane] = acc[r];
  }
  __syncthreads();
  {
    int r = tid >> 5, ln = tid & 31;
    int row = (ln < 16) ? r : r + 8;
    int cc = ln & 15;
    #pragma unroll
    for (int nt = 0; nt < 2; ++nt) {
      float v = part[nt * 256 + tid] + part[(nt + 2) * 256 + tid] +
                part[(nt + 4) * 256 + tid] + part[(nt + 6) * 256 + tid];
      resid[((long)(b * N_ + n0 + row)) * C_ + h * D_ + nt * 16 + cc] += v * rowinv[row];
    }
  }
}

// ---------------------------------------------------------------------------
// Static graph-edge masked attention (single head, C=64).
// Mask application fused with the row-max sweep.
// ---------------------------------------------------------------------------
__global__ __launch_bounds__(256) void static_attn_kernel(
    const _Float16* __restrict__ qs16, const _Float16* __restrict__ ks16,
    const _Float16* __restrict__ vsT16, const unsigned* __restrict__ maskbits,
    float* __restrict__ resid)
{
  extern __shared__ char smraw[];
  _Float16* sw  = (_Float16*)smraw;
  float* part   = (float*)(smraw + 16 * SWPITCH * 2);
  float* psum   = part + 8 * 256;
  float* pmax   = psum + 256;
  float* rowmax = pmax + 256;
  float* rowinv = rowmax + 32;

  int tid = threadIdx.x, lane = tid & 31, wv = tid >> 5;
  int tile = blockIdx.x & 127, b = blockIdx.x >> 7;
  int n0 = tile * 16;
  int m = lane & 15;

  // ---- Phase 1: scores (K=64 -> two WMMA per tile) ----
  long qbase = ((long)(b * N_ + n0 + m)) * C_;
  int aoff = (lane < 16) ? 0 : 16;
  int boff = (lane < 16) ? 0 : 16;
  v16h af0 = cat16(*(const v8h*)(qs16 + qbase + aoff),
                   *(const v8h*)(qs16 + qbase + aoff + 8));
  v16h af1 = cat16(*(const v8h*)(qs16 + qbase + 32 + aoff),
                   *(const v8h*)(qs16 + qbase + 32 + aoff + 8));
  for (int t = wv; t < 128; t += 8) {
    int c0 = t * 16;
    long kbase = ((long)(b * N_ + c0 + m)) * C_;
    v16h bf0 = cat16(*(const v8h*)(ks16 + kbase + boff),
                     *(const v8h*)(ks16 + kbase + boff + 8));
    v16h bf1 = cat16(*(const v8h*)(ks16 + kbase + 32 + boff),
                     *(const v8h*)(ks16 + kbase + 32 + boff + 8));
    v8f acc = {};
    acc = __builtin_amdgcn_wmma_f32_16x16x32_f16(false, af0, false, bf0, (short)0, acc, false, false);
    acc = __builtin_amdgcn_wmma_f32_16x16x32_f16(false, af1, false, bf1, (short)0, acc, false, false);
    #pragma unroll
    for (int r = 0; r < 8; ++r) {
      int row = (lane < 16) ? r : r + 8;
      sw[row * SWPITCH + c0 + m] = (_Float16)(acc[r] * SCALE);
    }
  }
  __syncthreads();

  // ---- Phase 2 (fused): apply edge mask + row max; vectorized ----
  int rowId = tid >> 4, sub = tid & 15;
  {
    int n = n0 + rowId;
    const unsigned* mrow = maskbits + n * 64;
    _Float16* rp = sw + rowId * SWPITCH + sub * 128;
    float mx = -3.0e38f;
    #pragma unroll 4
    for (int kk = 0; kk < 16; ++kk) {
      int k2 = (kk + sub) & 15;
      int col = sub * 128 + k2 * 8;
      unsigned sh = mrow[col >> 5] >> (col & 31);  // 8 mask bits (col mult of 8)
      v8h hv = *(const v8h*)(rp + k2 * 8);
      #pragma unroll
      for (int e = 0; e < 8; ++e) {
        if (!((sh >> e) & 1u)) hv[e] = (_Float16)NEGH;
        mx = fmaxf(mx, (float)hv[e]);
      }
      *(v8h*)(rp + k2 * 8) = hv;
    }
    pmax[tid] = mx;
  }
  __syncthreads();
  if (tid < 16) {
    float mx = -3.0e38f;
    for (int k2 = 0; k2 < 16; ++k2) mx = fmaxf(mx, pmax[tid * 16 + k2]);
    rowmax[tid] = mx;
  }
  __syncthreads();

  // ---- Phase 3: w = exp(s-max); denom ----
  {
    _Float16* rp = sw + rowId * SWPITCH + sub * 128;
    float mxv = rowmax[rowId], dn = 0.f;
    #pragma unroll 4
    for (int kk = 0; kk < 16; ++kk) {
      int k2 = (kk + sub) & 15;
      v8h hv = *(const v8h*)(rp + k2 * 8);
      v8h ov;
      #pragma unroll
      for (int e = 0; e < 8; ++e) {
        float wt = __expf((float)hv[e] - mxv);
        ov[e] = (_Float16)wt;
        dn += wt;
      }
      *(v8h*)(rp + k2 * 8) = ov;
    }
    psum[tid] = dn;
  }
  __syncthreads();
  if (tid < 16) {
    float dn = 0.f;
    for (int k2 = 0; k2 < 16; ++k2) dn += psum[tid * 16 + k2];
    rowinv[tid] = 1.0f / dn;
  }
  __syncthreads();

  // ---- Phase 4: out = (w * V)/denom over C=64 cols; resid += ----
  {
    int nt = wv & 3, kh = wv >> 2;
    int c0 = nt * 16;
    long vbase = ((long)(b * C_) + c0 + m) * N_;
    v8f acc = {};
    for (int kc = kh * 1024; kc < kh * 1024 + 1024; kc += 32) {
      v16h af = cat16(*(const v8h*)(sw + m * SWPITCH + kc + aoff),
                      *(const v8h*)(sw + m * SWPITCH + kc + aoff + 8));
      v16h bf = cat16(*(const v8h*)(vsT16 + vbase + kc + boff),
                      *(const v8h*)(vsT16 + vbase + kc + boff + 8));
      acc = __builtin_amdgcn_wmma_f32_16x16x32_f16(false, af, false, bf,
                                                   (short)0, acc, false, false);
    }
    #pragma unroll
    for (int r = 0; r < 8; ++r) part[wv * 256 + r * 32 + lane] = acc[r];
  }
  __syncthreads();
  {
    int r = tid >> 5, ln = tid & 31;
    int row = (ln < 16) ? r : r + 8;
    int cc = ln & 15;
    #pragma unroll
    for (int nt = 0; nt < 4; ++nt) {
      float v = part[nt * 256 + tid] + part[(nt + 4) * 256 + tid];
      resid[((long)(b * N_ + n0 + row)) * C_ + nt * 16 + cc] += v * rowinv[row];
    }
  }
}

// ---------------------------------------------------------------------------
// Fused LN1 -> MLP (64->128->64, no act) -> +residual -> LN2. One block/row.
// ---------------------------------------------------------------------------
static __device__ __forceinline__ float bsum64(float* red, int t, float v) {
  red[t] = v;
  __syncthreads();
  for (int s = 32; s > 0; s >>= 1) {
    if (t < s) red[t] += red[t + s];
    __syncthreads();
  }
  float r = red[0];
  __syncthreads();
  return r;
}

__global__ __launch_bounds__(64) void mlp_ln_kernel(
    const float* __restrict__ resid,
    const float* __restrict__ g1, const float* __restrict__ be1,
    const float* __restrict__ g2, const float* __restrict__ be2,
    const float* __restrict__ w1, const float* __restrict__ b1,
    const float* __restrict__ w2, const float* __restrict__ b2,
    float* __restrict__ out)
{
  __shared__ float red[64];
  __shared__ float lnr[64];
  __shared__ float mid[128];
  int t = threadIdx.x;
  long row = blockIdx.x;
  float rv = resid[row * 64 + t];

  float mean = bsum64(red, t, rv) * (1.f / 64.f);
  float d0 = rv - mean;
  float var = bsum64(red, t, d0 * d0) * (1.f / 64.f);
  lnr[t] = d0 * rsqrtf(var + 1e-5f) * g1[t] + be1[t];
  __syncthreads();

  float m1 = b1[t], m2 = b1[t + 64];
  #pragma unroll 8
  for (int c = 0; c < 64; ++c) {
    float l = lnr[c];
    m1 = fmaf(l, w1[c * 128 + t], m1);
    m2 = fmaf(l, w1[c * 128 + t + 64], m2);
  }
  mid[t] = m1; mid[t + 64] = m2;
  __syncthreads();

  float hs = b2[t];
  #pragma unroll 8
  for (int j = 0; j < 128; ++j) hs = fmaf(mid[j], w2[j * 64 + t], hs);
  float y = hs + rv;

  float mean2 = bsum64(red, t, y) * (1.f / 64.f);
  float d2 = y - mean2;
  float var2 = bsum64(red, t, d2 * d2) * (1.f / 64.f);
  out[row * 64 + t] = d2 * rsqrtf(var2 + 1e-5f) * g2[t] + be2[t];
}

// ---------------------------------------------------------------------------
extern "C" void kernel_launch(void* const* d_in, const int* in_sizes, int n_in,
                              void* d_out, int out_size, void* d_ws, size_t ws_size,
                              hipStream_t stream) {
  (void)in_sizes; (void)n_in; (void)out_size; (void)ws_size;
  const float* x     = (const float*)d_in[0];
  const int*   edge  = (const int*)d_in[1];
  // d_in[2] = num_heads (hardcoded H_=2)
  const float* wq_d  = (const float*)d_in[3];
  const float* wk_d  = (const float*)d_in[4];
  const float* wv_d  = (const float*)d_in[5];
  const float* bv_d  = (const float*)d_in[6];
  const float* wq_s  = (const float*)d_in[7];
  const float* wk_s  = (const float*)d_in[8];
  const float* wv_s  = (const float*)d_in[9];
  const float* bv_s  = (const float*)d_in[10];
  const float* ln1_g = (const float*)d_in[11];
  const float* ln1_b = (const float*)d_in[12];
  const float* ln2_g = (const float*)d_in[13];
  const float* ln2_b = (const float*)d_in[14];
  const float* w1    = (const float*)d_in[15];
  const float* b1    = (const float*)d_in[16];
  const float* w2    = (const float*)d_in[17];
  const float* b2    = (const float*)d_in[18];

  char* ws = (char*)d_ws;
  float*     resid    = (float*)ws;                       // 4 MB
  _Float16*  q16      = (_Float16*)(ws + 4194304);        // 2 MB each
  _Float16*  k16      = (_Float16*)(ws + 6291456);
  _Float16*  vT16     = (_Float16*)(ws + 8388608);
  _Float16*  qs16     = (_Float16*)(ws + 10485760);
  _Float16*  ks16     = (_Float16*)(ws + 12582912);
  _Float16*  vsT16    = (_Float16*)(ws + 14680064);
  unsigned*  maskbits = (unsigned*)(ws + 16777216);       // 512 KB

  float* out  = (float*)d_out;
  float* attn = out + (long)B_ * N_ * C_;                  // attn_d follows out

  size_t projSmem = (size_t)(6 * 4096 + 64 + 64 + 256) * sizeof(float);
  size_t attnSmem = (size_t)16 * SWPITCH * sizeof(_Float16)
                  + (size_t)(8 * 256 + 256 + 256 + 64) * sizeof(float);

  proj_kernel<<<256, 256, projSmem, stream>>>(
      x, wq_d, wk_d, wv_d, bv_d, wq_s, wk_s, wv_s, bv_s,
      resid, q16, k16, vT16, qs16, ks16, vsT16);

  mask_clear_kernel<<<512, 256, 0, stream>>>(maskbits);
  mask_set_kernel<<<E_ / 256, 256, 0, stream>>>(edge, maskbits);

  dyn_attn_kernel<<<B_ * H_ * (N_ / 16), 256, attnSmem, stream>>>(
      q16, k16, vT16, resid, attn);

  static_attn_kernel<<<B_ * (N_ / 16), 256, attnSmem, stream>>>(
      qs16, ks16, vsT16, maskbits, resid);

  mlp_ln_kernel<<<B_ * N_, 64, 0, stream>>>(
      resid, ln1_g, ln1_b, ln2_g, ln2_b, w1, b1, w2, b2, out);
}
